// Model_33346126086363
// MI455X (gfx1250) — compile-verified, matrix operand order checked by reading, not run on hardware
//
#include <hip/hip_runtime.h>
#include <hip/hip_bf16.h>
#include <math.h>

// ---------------------------------------------------------------------------
// Spatial attention (block-diagonal 25x25 frames) for MI455X / gfx1250.
// Dominant FLOPs (4x [6000,512]x[512,512] GEMMs) run on v_wmma_f32_16x16x32_f16
// with double-buffered LDS and software-pipelined global loads.
// Attention (25-token frames) is VALU per-frame (too small for 16x16 WMMA).
// Out-of-block logits are FLT_MIN*adj -> exp==0, so only the diagonal blocks
// of `adj` are read.
// ---------------------------------------------------------------------------

typedef __attribute__((ext_vector_type(16))) _Float16 v16h;
typedef __attribute__((ext_vector_type(8)))  _Float16 v8h;
typedef __attribute__((ext_vector_type(8)))  float    v8f;

#define SEQ_LEN  120
#define NJ       25
#define NTOK     3000            // SEQ_LEN * NJ
#define BS       2
#define MROWS    (BS * NTOK)     // 6000
#define DDIM     512
#define NHEADS   8
#define DHEAD    64

#define BM 128
#define BN 128
#define BK 32
#define KSTEPS (DDIM / BK)       // 16

// ---------------------------------------------------------------------------
// h = x + positional_encoding, converted to f16.  pe depends only on
// joint index (token % 25) and channel.
// ---------------------------------------------------------------------------
__global__ __launch_bounds__(256) void prep_h_kernel(const float* __restrict__ x,
                                                     _Float16* __restrict__ h) {
  int idx = blockIdx.x * 256 + threadIdx.x;
  if (idx >= MROWS * DDIM) return;
  int c     = idx & (DDIM - 1);
  int m     = idx >> 9;          // row index 0..5999
  int tok   = m % NTOK;
  int joint = tok % NJ;
  // div_term = exp((c & ~1) * (-ln(10000)/512))
  float div = __expf((float)(c & ~1) * (-9.210340371976184f / 512.0f));
  float ang = (float)joint * div;
  float pe  = (c & 1) ? __cosf(ang) : __sinf(ang);
  h[idx] = (_Float16)(x[idx] + pe);
}

// ---------------------------------------------------------------------------
// Transpose & convert the four 512x512 weights: Wt[w][n][k] = W[w][k][n]  (f16)
// so GEMM B-fragments are contiguous along K.
// ---------------------------------------------------------------------------
__global__ __launch_bounds__(256) void prep_wt_kernel(const float* __restrict__ Wq,
                                                      const float* __restrict__ Wk,
                                                      const float* __restrict__ Wv,
                                                      const float* __restrict__ Wo,
                                                      _Float16* __restrict__ Wt) {
  int idx = blockIdx.x * 256 + threadIdx.x;   // 4*512*512 total
  if (idx >= 4 * DDIM * DDIM) return;
  int w = idx >> 18;
  int n = (idx >> 9) & (DDIM - 1);
  int k = idx & (DDIM - 1);
  const float* W = (w == 0) ? Wq : (w == 1) ? Wk : (w == 2) ? Wv : Wo;
  Wt[idx] = (_Float16)W[k * DDIM + n];
}

// ---------------------------------------------------------------------------
// C[M,512] = A[M,512](f16) x Bt[512,512](f16, row n / col k) + bias (+ReLU)
// Workgroup: 256 threads = 8 waves; tile BM=128 x BN=128; each wave owns a
// 32x64 sub-tile = 2x4 WMMA fragments (8 v_wmma per K-step).  Double-buffered
// LDS: global loads for step s+1 issue before the 8 WMMAs of step s; the
// prefetched registers are stored to the alternate buffer after compute.
// ---------------------------------------------------------------------------
__global__ __launch_bounds__(256) void gemm_f16_wmma(const _Float16* __restrict__ A,
                                                     const _Float16* __restrict__ Bt,
                                                     const float* __restrict__ bias,
                                                     float* __restrict__ C,
                                                     int M, int relu) {
  __shared__ __align__(16) _Float16 Asl[2][BM * BK];   // 2 x 8 KB
  __shared__ __align__(16) _Float16 Bsl[2][BN * BK];   // 2 x 8 KB

  const int tid  = threadIdx.x;
  const int lane = tid & 31;
  const int wave = tid >> 5;        // 0..7
  const int wm   = wave & 3;        // 0..3  (row block of 32)
  const int wn   = wave >> 2;       // 0..1  (col block of 64)
  const int half = lane >> 4;       // K-group select within fragment
  const int l16  = lane & 15;       // M (A) / N (B) within fragment

  const int m0 = blockIdx.x * BM;
  const int n0 = blockIdx.y * BN;

  // per-thread staging slots: lin = p*256 + tid ; r = lin>>2 ; cg = (lin&3)*8
  const int st_r  = tid >> 2;            // rows covered by pass p: p*64 + st_r
  const int st_cg = (tid & 3) * 8;

  v8f acc[2][4];
#pragma unroll
  for (int i = 0; i < 2; ++i)
#pragma unroll
    for (int j = 0; j < 4; ++j) acc[i][j] = (v8f){0.f,0.f,0.f,0.f,0.f,0.f,0.f,0.f};

  uint4 ra[2], rb[2];

  auto load_tile = [&](int k0) {
#pragma unroll
    for (int p = 0; p < 2; ++p) {
      int r  = p * 64 + st_r;
      int gr = m0 + r;
      ra[p] = make_uint4(0u, 0u, 0u, 0u);
      if (gr < M) ra[p] = *(const uint4*)(A + (size_t)gr * DDIM + k0 + st_cg);
      rb[p] = *(const uint4*)(Bt + (size_t)(n0 + r) * DDIM + k0 + st_cg);
    }
  };
  auto store_tile = [&](int buf) {
#pragma unroll
    for (int p = 0; p < 2; ++p) {
      int r = p * 64 + st_r;
      *(uint4*)(&Asl[buf][r * BK + st_cg]) = ra[p];
      *(uint4*)(&Bsl[buf][r * BK + st_cg]) = rb[p];
    }
  };

  // prologue: stage K-step 0
  load_tile(0);
  store_tile(0);
  __syncthreads();

  for (int ks = 0; ks < KSTEPS; ++ks) {
    const int cur = ks & 1;

    // issue global loads for the next K-step (hidden under the WMMAs below)
    if (ks + 1 < KSTEPS) {
      load_tile((ks + 1) * BK);
      // pull the K-step-after-next into cache (global_prefetch_b8)
      if (ks + 2 < KSTEPS) {
        __builtin_prefetch(A + (size_t)(m0 + st_r) * DDIM + (ks + 2) * BK, 0, 1);
        __builtin_prefetch(Bt + (size_t)(n0 + st_r) * DDIM + (ks + 2) * BK, 0, 1);
      }
    }

    // assemble fragments from LDS (two ds_load_b128 each; ISA 16-bit layout:
    // lane holds row l16, K = half*8+{0..7} and 16+half*8+{0..7})
    v16h af[2], bf[4];
#pragma unroll
    for (int i = 0; i < 2; ++i) {
      const _Float16* base = &Asl[cur][(wm * 32 + i * 16 + l16) * BK + half * 8];
      ((v8h*)&af[i])[0] = *(const v8h*)(base);
      ((v8h*)&af[i])[1] = *(const v8h*)(base + 16);
    }
#pragma unroll
    for (int j = 0; j < 4; ++j) {
      const _Float16* base = &Bsl[cur][(wn * 64 + j * 16 + l16) * BK + half * 8];
      ((v8h*)&bf[j])[0] = *(const v8h*)(base);
      ((v8h*)&bf[j])[1] = *(const v8h*)(base + 16);
    }

#pragma unroll
    for (int i = 0; i < 2; ++i)
#pragma unroll
      for (int j = 0; j < 4; ++j)
        acc[i][j] = __builtin_amdgcn_wmma_f32_16x16x32_f16(
            /*neg_a=*/false, af[i], /*neg_b=*/false, bf[j],
            /*c_mod=*/(short)0, acc[i][j], /*reuse_a=*/false, /*reuse_b=*/false);

    // stage the prefetched tile into the alternate buffer
    if (ks + 1 < KSTEPS) store_tile(1 - cur);
    __syncthreads();
  }

  // --- epilogue: C layout VGPR g -> row g + half*8, col l16 ---
#pragma unroll
  for (int i = 0; i < 2; ++i) {
#pragma unroll
    for (int j = 0; j < 4; ++j) {
      int n = n0 + wn * 64 + j * 16 + l16;
      float bn = bias[n];
#pragma unroll
      for (int g = 0; g < 8; ++g) {
        int m = m0 + wm * 32 + i * 16 + half * 8 + g;
        if (m < M) {
          float v = acc[i][j][g] + bn;
          if (relu) v = fmaxf(v, 0.f);
          C[(size_t)m * DDIM + n] = v;
        }
      }
    }
  }
}

// ---------------------------------------------------------------------------
// Per-frame attention: one workgroup per (batch, frame), loops over 8 heads.
// S = (Q K^T / 8) * adj_block ; stable softmax with +1e-6 denominator ; P @ V.
// Writes f16 attention output [6000,512] (heads re-concatenated).
// ---------------------------------------------------------------------------
__global__ __launch_bounds__(256) void attention_kernel(const float* __restrict__ q,
                                                        const float* __restrict__ k,
                                                        const float* __restrict__ v,
                                                        const float* __restrict__ adj,
                                                        _Float16* __restrict__ attn) {
  __shared__ float Qs[NJ][DHEAD];
  __shared__ float Ks[NJ][DHEAD];
  __shared__ float Vs[NJ][DHEAD];
  __shared__ float S[NJ][NJ];
  __shared__ float Adj[NJ][NJ];

  const int bf = blockIdx.x;          // 0..239
  const int b  = bf / SEQ_LEN;
  const int f  = bf % SEQ_LEN;
  const int t0 = f * NJ;
  const int tid = threadIdx.x;

  // diagonal 25x25 block of adj for this (batch, frame)
  for (int e = tid; e < NJ * NJ; e += 256) {
    int i = e / NJ, j = e % NJ;
    Adj[i][j] = adj[(size_t)b * NTOK * NTOK + (size_t)(t0 + i) * NTOK + (t0 + j)];
  }

  for (int h = 0; h < NHEADS; ++h) {
    __syncthreads();
    for (int e = tid; e < NJ * DHEAD; e += 256) {
      int i = e >> 6, d = e & 63;
      size_t off = ((size_t)(b * NTOK + t0 + i)) * DDIM + h * DHEAD + d;
      Qs[i][d] = q[off];
      Ks[i][d] = k[off];
      Vs[i][d] = v[off];   // ReLU already applied in V-projection epilogue
    }
    __syncthreads();

    for (int e = tid; e < NJ * NJ; e += 256) {
      int i = e / NJ, j = e % NJ;
      float s = 0.f;
#pragma unroll
      for (int d = 0; d < DHEAD; ++d) s += Qs[i][d] * Ks[j][d];
      S[i][j] = s * 0.125f * Adj[i][j];   // 1/sqrt(64) * adjacency gate
    }
    __syncthreads();

    if (tid < NJ) {
      float mx = -3.0e38f;
#pragma unroll
      for (int j = 0; j < NJ; ++j) mx = fmaxf(mx, S[tid][j]);
      float sum = 0.f;
#pragma unroll
      for (int j = 0; j < NJ; ++j) {
        float e_ = __expf(S[tid][j] - mx);
        S[tid][j] = e_;
        sum += e_;
      }
      float inv = 1.0f / (sum + 1e-6f);
#pragma unroll
      for (int j = 0; j < NJ; ++j) S[tid][j] *= inv;
    }
    __syncthreads();

    for (int e = tid; e < NJ * DHEAD; e += 256) {
      int i = e >> 6, d = e & 63;
      float o = 0.f;
#pragma unroll
      for (int j = 0; j < NJ; ++j) o += S[i][j] * Vs[j][d];
      attn[((size_t)(b * NTOK + t0 + i)) * DDIM + h * DHEAD + d] = (_Float16)o;
    }
  }
}

// ---------------------------------------------------------------------------
// Row-wise LayerNorm (unbiased std, ddof=1): out = g*(y-mean)/(std+eps)+b
// ---------------------------------------------------------------------------
__global__ __launch_bounds__(256) void layernorm_kernel(const float* __restrict__ y,
                                                        const float* __restrict__ gamma,
                                                        const float* __restrict__ beta,
                                                        float* __restrict__ out) {
  const int row = blockIdx.x;
  const int tid = threadIdx.x;
  __shared__ float sh1[256];
  __shared__ float sh2[256];

  float s = 0.f, s2 = 0.f;
  for (int c = tid; c < DDIM; c += 256) {
    float t = y[(size_t)row * DDIM + c];
    s += t;
    s2 += t * t;
  }
  sh1[tid] = s;
  sh2[tid] = s2;
  __syncthreads();
  for (int off = 128; off > 0; off >>= 1) {
    if (tid < off) {
      sh1[tid] += sh1[tid + off];
      sh2[tid] += sh2[tid + off];
    }
    __syncthreads();
  }
  float mean = sh1[0] * (1.0f / DDIM);
  float var  = (sh2[0] - (float)DDIM * mean * mean) * (1.0f / (DDIM - 1));
  float inv  = 1.0f / (sqrtf(fmaxf(var, 0.f)) + 1e-6f);
  for (int c = tid; c < DDIM; c += 256) {
    float t = y[(size_t)row * DDIM + c];
    out[(size_t)row * DDIM + c] = gamma[c] * (t - mean) * inv + beta[c];
  }
}

// ---------------------------------------------------------------------------
// Launch
// ---------------------------------------------------------------------------
extern "C" void kernel_launch(void* const* d_in, const int* in_sizes, int n_in,
                              void* d_out, int out_size, void* d_ws, size_t ws_size,
                              hipStream_t stream) {
  const float* x    = (const float*)d_in[0];
  const float* adj  = (const float*)d_in[1];
  const float* Wq   = (const float*)d_in[2];
  const float* bq   = (const float*)d_in[3];
  const float* Wk   = (const float*)d_in[4];
  const float* bk   = (const float*)d_in[5];
  const float* Wv   = (const float*)d_in[6];
  const float* bv   = (const float*)d_in[7];
  const float* Wo   = (const float*)d_in[8];
  const float* bo   = (const float*)d_in[9];
  const float* ln_g = (const float*)d_in[10];
  const float* ln_b = (const float*)d_in[11];
  float* out = (float*)d_out;

  // workspace carve-up (all offsets 256B-aligned)
  char* ws = (char*)d_ws;
  size_t off = 0;
  auto take = [&](size_t bytes) {
    char* p = ws + off;
    off += (bytes + 255) & ~(size_t)255;
    return p;
  };
  _Float16* h_f16   = (_Float16*)take((size_t)MROWS * DDIM * 2);       //  6.1 MB
  _Float16* wt_f16  = (_Float16*)take((size_t)4 * DDIM * DDIM * 2);    //  2.1 MB
  float*    q_f32   = (float*)   take((size_t)MROWS * DDIM * 4);       // 12.3 MB
  float*    k_f32   = (float*)   take((size_t)MROWS * DDIM * 4);       // 12.3 MB
  float*    v_f32   = (float*)   take((size_t)MROWS * DDIM * 4);       // 12.3 MB
  _Float16* attn16  = (_Float16*)take((size_t)MROWS * DDIM * 2);       //  6.1 MB
  float*    y_f32   = (float*)   take((size_t)MROWS * DDIM * 4);       // 12.3 MB
  (void)ws_size; (void)in_sizes; (void)n_in; (void)out_size;

  // 1) h = x + PE  (f16)
  prep_h_kernel<<<(MROWS * DDIM + 255) / 256, 256, 0, stream>>>(x, h_f16);
  // 2) transpose weights to f16 [n][k]
  prep_wt_kernel<<<(4 * DDIM * DDIM + 255) / 256, 256, 0, stream>>>(Wq, Wk, Wv, Wo, wt_f16);

  // 3) WMMA projections
  dim3 gg((MROWS + BM - 1) / BM, DDIM / BN, 1);   // 47 x 4
  gemm_f16_wmma<<<gg, 256, 0, stream>>>(h_f16, wt_f16 + 0 * DDIM * DDIM, bq, q_f32, MROWS, 0);
  gemm_f16_wmma<<<gg, 256, 0, stream>>>(h_f16, wt_f16 + 1 * DDIM * DDIM, bk, k_f32, MROWS, 0);
  gemm_f16_wmma<<<gg, 256, 0, stream>>>(h_f16, wt_f16 + 2 * DDIM * DDIM, bv, v_f32, MROWS, 1);

  // 4) block-diagonal attention
  attention_kernel<<<BS * SEQ_LEN, 256, 0, stream>>>(q_f32, k_f32, v_f32, adj, attn16);

  // 5) output projection (+ReLU) on WMMA
  gemm_f16_wmma<<<gg, 256, 0, stream>>>(attn16, wt_f16 + 3 * DDIM * DDIM, bo, y_f32, MROWS, 1);

  // 6) LayerNorm -> d_out
  layernorm_kernel<<<MROWS, 256, 0, stream>>>(y_f32, ln_g, ln_b, out);
}
